// GPT_43696997270059
// MI455X (gfx1250) — matrix operand order repaired and checked by reading, not compiled, not run on hardware
//
#include <hip/hip_runtime.h>
#include <math.h>

// ---------------------------------------------------------------------------
// GPT forward for MI455X (gfx1250): bf16 WMMA GEMMs + flash attention.
// ---------------------------------------------------------------------------

typedef __bf16 bf16;
typedef __attribute__((ext_vector_type(16))) __bf16 v16bf;
typedef __attribute__((ext_vector_type(8)))  float  v8f;

#define TT 2048
#define CC 768
#define HH 12
#define HDM 64
#define LLY 4
#define FFD 3072
#define VOC 32000

#define WMMA_BF16(a, b, c) \
  __builtin_amdgcn_wmma_f32_16x16x32_bf16(false, (a), false, (b), (short)0, (c), false, false)

union FragBF { v16bf v; uint4 q[2]; };

__device__ __forceinline__ v8f zero8() {
  v8f z;
#pragma unroll
  for (int i = 0; i < 8; ++i) z[i] = 0.0f;
  return z;
}

// A-fragment (16x32 bf16, row-major source, ld in elements).
// lane L holds M = L%16; halves j=0..7 -> K = j + c0, j=8..15 -> K = j+8 + c0,
// c0 = 8*(L>=16). Two contiguous 16B loads per lane.
__device__ __forceinline__ v16bf load_frag_a(const bf16* __restrict__ p, int ld) {
  int lane = threadIdx.x & 31;
  const bf16* s = p + (lane & 15) * ld + ((lane >> 4) << 3);
  FragBF f;
  f.q[0] = *reinterpret_cast<const uint4*>(s);
  f.q[1] = *reinterpret_cast<const uint4*>(s + 16);
  return f.v;
}

// B-fragment (32x16 bf16) from K-major storage BT[n*ld + k].
// lane L holds N = L%16; halves j=0..7 -> K = j + c0, j=8..15 -> K = j+8 + c0,
// c0 = 8*(L>=16). Two contiguous 16B loads per lane.
__device__ __forceinline__ v16bf load_frag_bT(const bf16* __restrict__ p, int ld) {
  int lane = threadIdx.x & 31;
  const bf16* s = p + (lane & 15) * ld + ((lane >> 4) << 3);
  FragBF f;
  f.q[0] = *reinterpret_cast<const uint4*>(s);
  f.q[1] = *reinterpret_cast<const uint4*>(s + 16);
  return f.v;
}

// ---------------------------------------------------------------------------
// Weight conversion: fp32 -> bf16, stored transposed ([N][K]) for B-frags.
// ---------------------------------------------------------------------------
__global__ void k_cvt_t(const float* __restrict__ W, bf16* __restrict__ Bt,
                        int K, int N) {
  int idx = blockIdx.x * 256 + threadIdx.x;
  if (idx >= K * N) return;
  int n = idx / K;
  int kk = idx - n * K;
  Bt[idx] = (bf16)W[(size_t)kk * N + n];
}

// Wq/Wk/Wv layer slice: [H][C][HD] -> Bt[(h*64+d)*C + c]
__global__ void k_cvt_qkv(const float* __restrict__ W, bf16* __restrict__ Bt) {
  int idx = blockIdx.x * 256 + threadIdx.x;
  if (idx >= CC * CC) return;
  int n = idx / CC;
  int c = idx - n * CC;
  int hh = n >> 6;
  int d = n & 63;
  Bt[idx] = (bf16)W[((size_t)hh * CC + c) * HDM + d];
}

// ---------------------------------------------------------------------------
// Token + positional embedding: h[t,c] = emb[x[t],c] + pos[t,c]
// ---------------------------------------------------------------------------
__global__ void k_embed(const int* __restrict__ x, const float* __restrict__ emb,
                        const float* __restrict__ pos, float* __restrict__ h) {
  int idx = blockIdx.x * 256 + threadIdx.x;  // grid sized exactly T*C/256
  int t = idx / CC;
  int c = idx - t * CC;
  h[idx] = emb[(size_t)x[t] * CC + c] + pos[idx];
}

// ---------------------------------------------------------------------------
// LayerNorm (biased variance) -> bf16 output for GEMM A input
// ---------------------------------------------------------------------------
__global__ __launch_bounds__(256) void k_ln(const float* __restrict__ x,
                                            const float* __restrict__ g,
                                            const float* __restrict__ b,
                                            bf16* __restrict__ out) {
  int row = blockIdx.x;
  const float* xr = x + (size_t)row * CC;
  float s = 0.0f, s2 = 0.0f;
  for (int i = threadIdx.x; i < CC; i += 256) {
    float v = xr[i];
    s += v;
    s2 += v * v;
  }
#pragma unroll
  for (int m = 16; m >= 1; m >>= 1) {
    s += __shfl_xor(s, m, 32);
    s2 += __shfl_xor(s2, m, 32);
  }
  __shared__ float sb[16];
  int wid = threadIdx.x >> 5;
  if ((threadIdx.x & 31) == 0) { sb[wid] = s; sb[8 + wid] = s2; }
  __syncthreads();
  s = 0.0f; s2 = 0.0f;
#pragma unroll
  for (int i = 0; i < 8; ++i) { s += sb[i]; s2 += sb[8 + i]; }
  float mean = s * (1.0f / CC);
  float var = s2 * (1.0f / CC) - mean * mean;
  float inv = rsqrtf(var + 1e-5f);
  for (int i = threadIdx.x; i < CC; i += 256)
    out[(size_t)row * CC + i] = (bf16)((xr[i] - mean) * inv * g[i] + b[i]);
}

// ---------------------------------------------------------------------------
// Fused GEMM: C = A(bf16 MxK, row-major) * B (bf16, stored [N][K]) [+bias]
//             [+resid] [gelu] -> f32 and/or bf16 (optionally transposed)
// Block: 256 thr = 8 waves; wave tile 32x32; block tile 64 x 128.
// ---------------------------------------------------------------------------
__global__ __launch_bounds__(256) void k_gemm(
    const bf16* __restrict__ A, const bf16* __restrict__ Bt,
    float* __restrict__ Cf, bf16* __restrict__ Cbf, int cbfT,
    const float* __restrict__ bias, const float* __restrict__ resid,
    int act, int M, int N, int K) {
  int wid = threadIdx.x >> 5;
  int lane = threadIdx.x & 31;
  int wm = wid & 1, wn = wid >> 1;
  int mb = blockIdx.y * 64 + wm * 32;
  int nb = blockIdx.x * 128 + wn * 32;

  v8f acc[2][2];
  acc[0][0] = zero8(); acc[0][1] = zero8();
  acc[1][0] = zero8(); acc[1][1] = zero8();

  const bf16* Ar0 = A + (size_t)mb * K;
  const bf16* Ar1 = A + (size_t)(mb + 16) * K;
  const bf16* Bc0 = Bt + (size_t)nb * K;
  const bf16* Bc1 = Bt + (size_t)(nb + 16) * K;

  for (int kb = 0; kb < K; kb += 32) {
    v16bf a0 = load_frag_a(Ar0 + kb, K);
    v16bf a1 = load_frag_a(Ar1 + kb, K);
    v16bf b0 = load_frag_bT(Bc0 + kb, K);
    v16bf b1 = load_frag_bT(Bc1 + kb, K);
    acc[0][0] = WMMA_BF16(a0, b0, acc[0][0]);
    acc[0][1] = WMMA_BF16(a0, b1, acc[0][1]);
    acc[1][0] = WMMA_BF16(a1, b0, acc[1][0]);
    acc[1][1] = WMMA_BF16(a1, b1, acc[1][1]);
  }

  int colx = lane & 15;
  int rowoff = (lane >> 4) << 3;
#pragma unroll
  for (int i = 0; i < 2; ++i)
#pragma unroll
    for (int j = 0; j < 2; ++j) {
      int col = nb + j * 16 + colx;
      float bv = bias ? bias[col] : 0.0f;
#pragma unroll
      for (int r = 0; r < 8; ++r) {
        int row = mb + i * 16 + rowoff + r;
        float val = acc[i][j][r] + bv;
        if (resid) val += resid[(size_t)row * N + col];
        if (act) val = 0.5f * val * (1.0f + erff(val * 0.70710678118f));
        if (Cf) Cf[(size_t)row * N + col] = val;
        if (Cbf) {
          if (cbfT) Cbf[(size_t)col * M + row] = (bf16)val;
          else      Cbf[(size_t)row * N + col] = (bf16)val;
        }
      }
    }
}

// ---------------------------------------------------------------------------
// Causal flash attention. One wave per (head, 16-query tile).
// q,k: bf16 [T][C] (head slice h*64..); vT: bf16 [C][T]; o: bf16 [T][C].
// ---------------------------------------------------------------------------
__global__ __launch_bounds__(128) void k_attn(const bf16* __restrict__ q,
                                              const bf16* __restrict__ k,
                                              const bf16* __restrict__ vT,
                                              bf16* __restrict__ o) {
  __shared__ __align__(16) bf16 psh[4][16 * 32];
  int wid = threadIdx.x >> 5;
  int lane = threadIdx.x & 31;
  int gw = blockIdx.x * 4 + wid;  // 0 .. H*(T/16)-1
  int head = gw >> 7;             // / (T/16 = 128)
  int qt = gw & 127;
  int q0 = qt << 4;

  v16bf qa0 = load_frag_a(q + (size_t)q0 * CC + head * HDM, CC);
  v16bf qa1 = load_frag_a(q + (size_t)q0 * CC + head * HDM + 32, CC);

  float mrow[8], lrow[8];
#pragma unroll
  for (int r = 0; r < 8; ++r) { mrow[r] = -1e30f; lrow[r] = 0.0f; }
  v8f oacc[4];
#pragma unroll
  for (int g = 0; g < 4; ++g) oacc[g] = zero8();

  int colx = lane & 15;
  int rowoff = (lane >> 4) << 3;
  bf16* pw = psh[wid];

  for (int kc = 0; kc < q0 + 16; kc += 32) {
    // --- scores: S = (q . k^T) for two 16-key tiles -----------------------
    v8f s0 = zero8(), s1 = zero8();
    {
      v16bf kb0 = load_frag_bT(k + (size_t)kc * CC + head * HDM, CC);
      v16bf kb1 = load_frag_bT(k + (size_t)kc * CC + head * HDM + 32, CC);
      s0 = WMMA_BF16(qa0, kb0, s0);
      s0 = WMMA_BF16(qa1, kb1, s0);
    }
    if (kc < q0) {  // wave-uniform: second tile has at least one valid key
      v16bf kb0 = load_frag_bT(k + (size_t)(kc + 16) * CC + head * HDM, CC);
      v16bf kb1 = load_frag_bT(k + (size_t)(kc + 16) * CC + head * HDM + 32, CC);
      s1 = WMMA_BF16(qa0, kb0, s1);
      s1 = WMMA_BF16(qa1, kb1, s1);
    }

    // --- online softmax (rows live across lanes 0-15 / 16-31) ------------
    float p0[8], p1[8];
#pragma unroll
    for (int r = 0; r < 8; ++r) {
      int row = q0 + rowoff + r;
      int c0 = kc + colx;
      int c1 = kc + 16 + colx;
      float v0 = (c0 <= row) ? s0[r] * 0.125f : -1e30f;  // 1/sqrt(64)
      float v1 = (c1 <= row) ? s1[r] * 0.125f : -1e30f;
      float rm = fmaxf(v0, v1);
#pragma unroll
      for (int ms = 1; ms < 16; ms <<= 1) rm = fmaxf(rm, __shfl_xor(rm, ms, 32));
      float mn = fmaxf(mrow[r], rm);
      float alpha = __expf(mrow[r] - mn);
      float e0 = __expf(v0 - mn);
      float e1 = __expf(v1 - mn);
      float rs = e0 + e1;
#pragma unroll
      for (int ms = 1; ms < 16; ms <<= 1) rs += __shfl_xor(rs, ms, 32);
      lrow[r] = lrow[r] * alpha + rs;
      mrow[r] = mn;
#pragma unroll
      for (int g = 0; g < 4; ++g) oacc[g][r] *= alpha;
      p0[r] = e0;
      p1[r] = e1;
    }

    // --- reshape P (C/D layout) -> A layout through LDS -------------------
#pragma unroll
    for (int r = 0; r < 8; ++r) {
      int prow = rowoff + r;
      pw[prow * 32 + colx]      = (bf16)p0[r];
      pw[prow * 32 + 16 + colx] = (bf16)p1[r];
    }
    asm volatile("s_wait_dscnt 0" ::: "memory");
    v16bf pa = load_frag_a(pw, 32);

    // --- O += P . V  (V^T storage -> contiguous B-frag loads) -------------
#pragma unroll
    for (int g = 0; g < 4; ++g) {
      v16bf vb = load_frag_bT(vT + (size_t)(head * HDM + g * 16) * TT + kc, TT);
      oacc[g] = WMMA_BF16(pa, vb, oacc[g]);
    }
  }

#pragma unroll
  for (int g = 0; g < 4; ++g)
#pragma unroll
    for (int r = 0; r < 8; ++r) {
      int t = q0 + rowoff + r;
      int d = head * HDM + g * 16 + colx;
      o[(size_t)t * CC + d] = (bf16)(oacc[g][r] / lrow[r]);
    }
}

// ---------------------------------------------------------------------------
// Host orchestration
// ---------------------------------------------------------------------------
static inline void launch_gemm(const bf16* A, const bf16* Bt, float* Cf, bf16* Cbf,
                               int cbfT, const float* bias, const float* resid,
                               int act, int M, int N, int K, hipStream_t s) {
  dim3 grid(N / 128, M / 64);
  k_gemm<<<grid, 256, 0, s>>>(A, Bt, Cf, Cbf, cbfT, bias, resid, act, M, N, K);
}

extern "C" void kernel_launch(void* const* d_in, const int* in_sizes, int n_in,
                              void* d_out, int out_size, void* d_ws, size_t ws_size,
                              hipStream_t stream) {
  const int*   x     = (const int*)d_in[0];
  const float* emb   = (const float*)d_in[1];
  const float* pos   = (const float*)d_in[2];
  const float* Wq    = (const float*)d_in[3];
  const float* Wk    = (const float*)d_in[4];
  const float* Wv    = (const float*)d_in[5];
  const float* Wproj = (const float*)d_in[6];
  const float* bproj = (const float*)d_in[7];
  const float* Wff   = (const float*)d_in[8];
  const float* bff   = (const float*)d_in[9];
  const float* Wffp  = (const float*)d_in[10];
  const float* bffp  = (const float*)d_in[11];
  const float* ln1g  = (const float*)d_in[12];
  const float* ln1b  = (const float*)d_in[13];
  const float* ln2g  = (const float*)d_in[14];
  const float* ln2b  = (const float*)d_in[15];
  const float* lnfg  = (const float*)d_in[16];
  const float* lnfb  = (const float*)d_in[17];
  const float* Wout  = (const float*)d_in[18];
  const float* bout  = (const float*)d_in[19];

  // ---- workspace carve-up (~140 MB total) --------------------------------
  char* wp = (char*)d_ws;
  auto take = [&](size_t bytes) -> char* {
    char* p = wp;
    wp += (bytes + 255) & ~(size_t)255;
    return p;
  };
  const size_t CCxCC = (size_t)CC * CC;
  const size_t CxF = (size_t)CC * FFD;
  bf16* wq_t   = (bf16*)take(sizeof(bf16) * LLY * CCxCC);
  bf16* wk_t   = (bf16*)take(sizeof(bf16) * LLY * CCxCC);
  bf16* wv_t   = (bf16*)take(sizeof(bf16) * LLY * CCxCC);
  bf16* wp_t   = (bf16*)take(sizeof(bf16) * LLY * CCxCC);
  bf16* wff_t  = (bf16*)take(sizeof(bf16) * LLY * CxF);
  bf16* wffp_t = (bf16*)take(sizeof(bf16) * LLY * CxF);
  bf16* wout_t = (bf16*)take(sizeof(bf16) * (size_t)VOC * CC);
  float* h     = (float*)take(sizeof(float) * (size_t)TT * CC);
  bf16* lnbuf  = (bf16*)take(sizeof(bf16) * (size_t)TT * CC);
  bf16* qb     = (bf16*)take(sizeof(bf16) * (size_t)TT * CC);
  bf16* kb     = (bf16*)take(sizeof(bf16) * (size_t)TT * CC);
  bf16* vTb    = (bf16*)take(sizeof(bf16) * (size_t)TT * CC);
  bf16* ob     = (bf16*)take(sizeof(bf16) * (size_t)TT * CC);
  bf16* ffb    = (bf16*)take(sizeof(bf16) * (size_t)TT * FFD);

  // ---- weight conversion (deterministic, every call) --------------------
  const int gQKV = (CC * CC + 255) / 256;
  for (int l = 0; l < LLY; ++l) {
    k_cvt_qkv<<<gQKV, 256, 0, stream>>>(Wq + (size_t)l * CCxCC, wq_t + (size_t)l * CCxCC);
    k_cvt_qkv<<<gQKV, 256, 0, stream>>>(Wk + (size_t)l * CCxCC, wk_t + (size_t)l * CCxCC);
    k_cvt_qkv<<<gQKV, 256, 0, stream>>>(Wv + (size_t)l * CCxCC, wv_t + (size_t)l * CCxCC);
    k_cvt_t<<<gQKV, 256, 0, stream>>>(Wproj + (size_t)l * CCxCC, wp_t + (size_t)l * CCxCC, CC, CC);
    int gF = ((int)CxF + 255) / 256;
    k_cvt_t<<<gF, 256, 0, stream>>>(Wff + (size_t)l * CxF, wff_t + (size_t)l * CxF, CC, FFD);
    k_cvt_t<<<gF, 256, 0, stream>>>(Wffp + (size_t)l * CxF, wffp_t + (size_t)l * CxF, FFD, CC);
  }
  k_cvt_t<<<(CC * VOC + 255) / 256, 256, 0, stream>>>(Wout, wout_t, CC, VOC);

  // ---- embedding --------------------------------------------------------
  k_embed<<<(TT * CC) / 256, 256, 0, stream>>>(x, emb, pos, h);

  // ---- transformer layers -----------------------------------------------
  const int attnBlocks = (HH * (TT / 16)) / 4;  // 4 waves per block
  for (int l = 0; l < LLY; ++l) {
    k_ln<<<TT, 256, 0, stream>>>(h, ln1g + l * CC, ln1b + l * CC, lnbuf);
    launch_gemm(lnbuf, wq_t + (size_t)l * CCxCC, nullptr, qb, 0, nullptr, nullptr, 0, TT, CC, CC, stream);
    launch_gemm(lnbuf, wk_t + (size_t)l * CCxCC, nullptr, kb, 0, nullptr, nullptr, 0, TT, CC, CC, stream);
    launch_gemm(lnbuf, wv_t + (size_t)l * CCxCC, nullptr, vTb, 1, nullptr, nullptr, 0, TT, CC, CC, stream);
    k_attn<<<attnBlocks, 128, 0, stream>>>(qb, kb, vTb, ob);
    launch_gemm(ob, wp_t + (size_t)l * CCxCC, h, nullptr, 0, bproj + l * CC, h, 0, TT, CC, CC, stream);
    k_ln<<<TT, 256, 0, stream>>>(h, ln2g + l * CC, ln2b + l * CC, lnbuf);
    launch_gemm(lnbuf, wff_t + (size_t)l * CxF, nullptr, ffb, 0, bff + l * FFD, nullptr, 1, TT, FFD, CC, stream);
    launch_gemm(ffb, wffp_t + (size_t)l * CxF, h, nullptr, 0, bffp + l * CC, h, 0, TT, CC, FFD, stream);
  }

  // ---- final LN + logits -------------------------------------------------
  k_ln<<<TT, 256, 0, stream>>>(h, lnfg, lnfb, lnbuf);
  launch_gemm(lnbuf, wout_t, (float*)d_out, nullptr, 0, bout, nullptr, 0, TT, VOC, CC, stream);
}
